// LengthRegulator_37409165148552
// MI455X (gfx1250) — matrix operand order repaired and checked by reading, not compiled, not run on hardware
//
#include <hip/hip_runtime.h>

// ---------------- types ----------------
typedef __attribute__((ext_vector_type(16))) __bf16 v16bf;
typedef __attribute__((ext_vector_type(8)))  float  v8f;

#define DMODEL 512
#define TLEN   1024
#define NBATCH 32
#define LMAX   4096

// round-to-nearest-even f32 -> bf16
__device__ __forceinline__ unsigned short f2bf(float f) {
    unsigned int u = __float_as_uint(f);
    u += 0x7FFFu + ((u >> 16) & 1u);
    return (unsigned short)(u >> 16);
}

// ---------------------------------------------------------------------------
// Repack conv weight [3][512][512] (HIO, f32) into per-lane bf16 B-fragments.
// Fragment layout (V_WMMA_F32_16X16X32_BF16, B = 32x16 K-major):
//   lane&15 = column N within the 16-wide tile
//   lanes 0-15 hold K = k0..k0+7, k0+16..k0+23 (2 per VGPR), lanes 16-31 hold K = k0+8..k0+15, k0+24..k0+31
// wpack index: (((kstep*32 + ntile)*32 + lane)*16 + j) bf16
// ---------------------------------------------------------------------------
__global__ void repack_w(const float* __restrict__ w, unsigned short* __restrict__ wp) {
    int tid  = blockIdx.x * 256 + threadIdx.x;   // 48*32*32 = 49152 threads
    int lane = tid & 31;
    int nt   = (tid >> 5) & 31;
    int ks   = tid >> 10;                        // 0..47
    int col   = nt * 16 + (lane & 15);
    int kbase = ks * 32 + ((lane >> 4) << 3);    // +8 for upper lane half
    unsigned short tmp[16];
#pragma unroll
    for (int j = 0; j < 16; ++j) {
        int kg = kbase + (j & 7) + ((j >> 3) << 4); // im2col K index in [0,1536)
        int sh = kg >> 9;                        // conv tap 0..2
        int ki = kg & 511;                       // input channel
        tmp[j] = f2bf(w[((size_t)sh * 512 + ki) * 512 + col]);
    }
    uint4* dst = (uint4*)&wp[(size_t)tid * 16];
    const uint4* src = (const uint4*)tmp;
    dst[0] = src[0];
    dst[1] = src[1];
}

// ---------------------------------------------------------------------------
// Fused conv1d(k=3,SAME) GEMM (bf16 WMMA) + bias + ReLU + LayerNorm.
// One block = 32 rows (two 16-row sub-tiles) x full 512 cols.
// 8 waves, each wave owns 64 columns: 2x4 accumulators of 16x16 f32.
// Each B fragment feeds two WMMAs (both row sub-tiles) -> halves L2 weight
// traffic vs a 16-row tile.
// A-tile (bf16, +halo) and the f32 LayerNorm stage alias one LDS buffer;
// the LN epilogue runs in two 16-row phases after the A-tile is dead.
// IN_F32: input is f32 (encoder) else bf16 (h1).
// DO_LINEAR: instead of storing h (bf16), compute dot(h_row, lin_w)+lin_b -> dpo.
// ---------------------------------------------------------------------------
#define AT_STRIDE  520            // ushorts per A-tile row (16B-aligned, bank-spread)
#define ST_STRIDE  516            // floats per stage row (bank-spread)
#define SMEM_BYTES 35360          // max(34*520*2, 16*516*4)

template <bool IN_F32, bool DO_LINEAR>
__global__ __launch_bounds__(256)
void conv_ln_kernel(const void* __restrict__ in,
                    const unsigned short* __restrict__ wp,
                    const float* __restrict__ bias,
                    const float* __restrict__ gamma,
                    const float* __restrict__ beta,
                    unsigned short* __restrict__ h_out,
                    const float* __restrict__ linw,
                    const float* __restrict__ linb,
                    float* __restrict__ dpo) {
    __shared__ __align__(16) unsigned char smem[SMEM_BYTES];
    unsigned short* At   = (unsigned short*)smem;   // 34 x 520 bf16 (GEMM phase)
    float*          stage = (float*)smem;           // 16 x 516 f32  (LN phase)

    const int tid  = threadIdx.x;
    const int lane = tid & 31;
    const int wv   = tid >> 5;                  // wave 0..7
    const int mt   = blockIdx.x;                // 0..1023
    const int b    = mt >> 5;                   // 32 row-tiles per batch
    const int t0   = (mt & 31) << 5;            // 32 rows per tile

    // ---- load A tile (rows t0-1 .. t0+32 -> 34 rows) into LDS as bf16 ----
    if (IN_F32) {
        const float* src = (const float*)in;
        for (int c = tid; c < 34 * 128; c += 256) {          // 128 float4 per row
            int row = c >> 7, f4 = c & 127;
            int t = t0 + row - 1;
            float4 v = make_float4(0.f, 0.f, 0.f, 0.f);
            if (t >= 0 && t < TLEN)
                v = ((const float4*)(src + ((size_t)b * TLEN + t) * DMODEL))[f4];
            unsigned long long pk =
                 (unsigned long long)f2bf(v.x)
               | ((unsigned long long)f2bf(v.y) << 16)
               | ((unsigned long long)f2bf(v.z) << 32)
               | ((unsigned long long)f2bf(v.w) << 48);
            *(unsigned long long*)&At[row * AT_STRIDE + f4 * 4] = pk;
        }
    } else {
        const unsigned short* src = (const unsigned short*)in;
        for (int c = tid; c < 34 * 64; c += 256) {           // 64 uint4 per row
            int row = c >> 6, q = c & 63;
            int t = t0 + row - 1;
            uint4 v = make_uint4(0u, 0u, 0u, 0u);
            if (t >= 0 && t < TLEN)
                v = ((const uint4*)(src + ((size_t)b * TLEN + t) * DMODEL))[q];
            ((uint4*)&At[row * AT_STRIDE])[q] = v;
        }
    }
    __syncthreads();

    // ---- GEMM: K = 1536 in 48 steps of 32, bf16 WMMA ----
    v8f acc[2][4];
#pragma unroll
    for (int rt = 0; rt < 2; ++rt)
#pragma unroll
        for (int nt = 0; nt < 4; ++nt)
#pragma unroll
            for (int v = 0; v < 8; ++v) acc[rt][nt][v] = 0.f;

    const int half8 = (lane >> 4) << 3;   // K sub-offset per A layout (0 / 8)
    const int arow  = lane & 15;

    for (int kk = 0; kk < 48; ++kk) {
        const int shift = kk >> 4;                 // conv tap
        const int kin   = (kk & 15) << 5;          // channel offset (x32)
        union { v16bf v; uint4 q[2]; } A0, A1;
        const unsigned short* ap0 = &At[(arow +      shift) * AT_STRIDE + kin + half8];
        const unsigned short* ap1 = &At[(arow + 16 + shift) * AT_STRIDE + kin + half8];
        A0.q[0] = *(const uint4*)ap0;              // VGPR0-3: K = kin+half8 .. +7
        A0.q[1] = *(const uint4*)(ap0 + 16);       // VGPR4-7: K = kin+16+half8 .. +7
        A1.q[0] = *(const uint4*)ap1;
        A1.q[1] = *(const uint4*)(ap1 + 16);
#pragma unroll
        for (int nt = 0; nt < 4; ++nt) {
            union { v16bf v; uint4 q[2]; } Bf;
            const uint4* bp = (const uint4*)
                &wp[(((size_t)kk * 32 + (wv * 4 + nt)) * 32 + lane) * 16];
            Bf.q[0] = bp[0];
            Bf.q[1] = bp[1];
            acc[0][nt] = __builtin_amdgcn_wmma_f32_16x16x32_bf16(
                false, A0.v, false, Bf.v, (short)0, acc[0][nt], false, false);
            acc[1][nt] = __builtin_amdgcn_wmma_f32_16x16x32_bf16(
                false, A1.v, false, Bf.v, (short)0, acc[1][nt], false, false);
        }
    }
    __syncthreads();   // A-tile dead; LDS becomes the f32 stage

    // ---- epilogue in two 16-row phases: bias+ReLU -> stage -> LayerNorm ----
#pragma unroll
    for (int rt = 0; rt < 2; ++rt) {
        // bias + ReLU into stage (C layout: row = v + 8*(lane>=16))
#pragma unroll
        for (int nt = 0; nt < 4; ++nt) {
            int col = wv * 64 + nt * 16 + (lane & 15);
            float bv = bias[col];
#pragma unroll
            for (int v = 0; v < 8; ++v) {
                float y = acc[rt][nt][v] + bv;
                stage[(v + half8) * ST_STRIDE + col] = fmaxf(y, 0.f);
            }
        }
        __syncthreads();

        // LayerNorm per row (wave-wide), fused epilogue
        for (int rr = 0; rr < 2; ++rr) {
            int r = wv * 2 + rr;
            float x[16];
            float s = 0.f, s2 = 0.f;
#pragma unroll
            for (int k = 0; k < 16; ++k) {
                float v = stage[r * ST_STRIDE + lane + 32 * k];
                x[k] = v; s += v; s2 += v * v;
            }
#pragma unroll
            for (int off = 16; off >= 1; off >>= 1) {
                s  += __shfl_xor(s,  off, 32);
                s2 += __shfl_xor(s2, off, 32);
            }
            float mu  = s * (1.f / 512.f);
            float var = s2 * (1.f / 512.f) - mu * mu;
            float rs  = rsqrtf(var + 1e-5f);
            size_t grow = (size_t)mt * 32 + rt * 16 + r;   // == b*1024 + t0 + rt*16 + r
            if (DO_LINEAR) {
                float dot = 0.f;
#pragma unroll
                for (int k = 0; k < 16; ++k) {
                    int col = lane + 32 * k;
                    float y = (x[k] - mu) * rs * gamma[col] + beta[col];
                    dot += y * linw[col];
                }
#pragma unroll
                for (int off = 16; off >= 1; off >>= 1) dot += __shfl_xor(dot, off, 32);
                if (lane == 0) dpo[grow] = dot + linb[0];
            } else {
                unsigned short* hp = h_out + grow * DMODEL;
#pragma unroll
                for (int k = 0; k < 16; ++k) {
                    int col = lane + 32 * k;
                    float y = (x[k] - mu) * rs * gamma[col] + beta[col];
                    hp[col] = f2bf(y);
                }
            }
        }
        __syncthreads();   // stage reads done before next phase overwrites
    }
}

// ---------------------------------------------------------------------------
// Inclusive cumsum of durations: one 1024-thread block per batch row.
// ---------------------------------------------------------------------------
__global__ __launch_bounds__(1024)
void cumsum_kernel(const int* __restrict__ dur, int* __restrict__ cum) {
    __shared__ int s[1024];
    int b = blockIdx.x, tid = threadIdx.x;
    s[tid] = dur[b * TLEN + tid];
    __syncthreads();
    for (int off = 1; off < 1024; off <<= 1) {
        int v = (tid >= off) ? s[tid - off] : 0;
        __syncthreads();
        s[tid] += v;
        __syncthreads();
    }
    cum[b * TLEN + tid] = s[tid];
}

// ---------------------------------------------------------------------------
// Length regulation: per block, 16 output frames of one batch.
// Wave-per-frame: uniform binary search over LDS cumsum, then float4 row copy.
// ---------------------------------------------------------------------------
__global__ __launch_bounds__(256)
void expand_kernel(const float* __restrict__ enc, const int* __restrict__ cum,
                   float* __restrict__ out) {
    __shared__ int c[1024];
    int blk = blockIdx.x;            // 32 batches * 256 blocks
    int b   = blk >> 8;
    int j0  = (blk & 255) << 4;
    int tid = threadIdx.x, lane = tid & 31, wv = tid >> 5;
    for (int i = tid; i < 1024; i += 256) c[i] = cum[b * TLEN + i];
    __syncthreads();
    int total = c[1023];
    for (int rr = 0; rr < 2; ++rr) {
        int j = j0 + wv * 2 + rr;
        float4* dst = (float4*)(out + ((size_t)b * LMAX + j) * DMODEL);
        if (j < total) {
            int lo = 0, hi = 1024;                       // searchsorted 'right'
            while (lo < hi) { int mid = (lo + hi) >> 1;
                              if (c[mid] <= j) lo = mid + 1; else hi = mid; }
            int idx = lo < 1023 ? lo : 1023;
            const float4* srcp = (const float4*)(enc + ((size_t)b * TLEN + idx) * DMODEL);
#pragma unroll
            for (int it = 0; it < 4; ++it) dst[it * 32 + lane] = srcp[it * 32 + lane];
        } else {
            float4 z = make_float4(0.f, 0.f, 0.f, 0.f);
#pragma unroll
            for (int it = 0; it < 4; ++it) dst[it * 32 + lane] = z;
        }
    }
}

// ---------------------------------------------------------------------------
extern "C" void kernel_launch(void* const* d_in, const int* in_sizes, int n_in,
                              void* d_out, int out_size, void* d_ws, size_t ws_size,
                              hipStream_t stream) {
    const float* enc = (const float*)d_in[0];
    const int*   dur = (const int*)d_in[1];
    const float* w1  = (const float*)d_in[2];
    const float* b1  = (const float*)d_in[3];
    const float* g1  = (const float*)d_in[4];
    const float* be1 = (const float*)d_in[5];
    const float* w2  = (const float*)d_in[6];
    const float* b2  = (const float*)d_in[7];
    const float* g2  = (const float*)d_in[8];
    const float* be2 = (const float*)d_in[9];
    const float* lw  = (const float*)d_in[10];
    const float* lb  = (const float*)d_in[11];

    float* out = (float*)d_out;                                    // expanded [32,4096,512]
    float* dpo = out + (size_t)NBATCH * LMAX * DMODEL;             // dpo [32,1024]

    // workspace: repacked weights + cumsum (~3.3 MB)
    unsigned char* ws = (unsigned char*)d_ws;
    unsigned short* wp1 = (unsigned short*)ws;                     // 1,572,864 B
    unsigned short* wp2 = (unsigned short*)(ws + 1572864);         // 1,572,864 B
    int*            cum = (int*)(ws + 3145728);                    //   131,072 B

    // h1 (bf16, 32 MB) parked at the front of d_out's expanded region;
    // the expand kernel fully overwrites it afterwards (stream-ordered).
    unsigned short* h1 = (unsigned short*)d_out;

    repack_w<<<192, 256, 0, stream>>>(w1, wp1);
    repack_w<<<192, 256, 0, stream>>>(w2, wp2);
    conv_ln_kernel<true,  false><<<1024, 256, 0, stream>>>(
        enc, wp1, b1, g1, be1, h1, nullptr, nullptr, nullptr);
    conv_ln_kernel<false, true ><<<1024, 256, 0, stream>>>(
        h1,  wp2, b2, g2, be2, nullptr, lw, lb, dpo);
    cumsum_kernel<<<NBATCH, 1024, 0, stream>>>(dur, cum);
    expand_kernel<<<NBATCH * 256, 256, 0, stream>>>(enc, cum, out);
}
